// MultiHeadSelfAttention_22789096472791
// MI455X (gfx1250) — compile-verified, matrix operand order checked by reading, not run on hardware
//
#include <hip/hip_runtime.h>

typedef unsigned short u16;
typedef unsigned int   u32;
typedef unsigned long long u64;

typedef __attribute__((ext_vector_type(16))) __bf16 v16bf;
typedef __attribute__((ext_vector_type(8)))  float  v8f;

union BF16Frag { uint4 u4[2]; v16bf v; };

__device__ __forceinline__ u16 f2bf(float f) {
  u32 u = __float_as_uint(f);
  u = (u + 0x7FFFu + ((u >> 16) & 1u)) >> 16;  // round-to-nearest-even
  return (u16)u;
}
__device__ __forceinline__ float bf2f(u16 h) {
  return __uint_as_float(((u32)h) << 16);
}

// ---------------------------------------------------------------------------
// fp32 -> bf16 cast
// ---------------------------------------------------------------------------
__global__ void cast_f32_to_bf16(const float* __restrict__ in, u16* __restrict__ out, size_t n) {
  size_t i = (size_t)blockIdx.x * blockDim.x + threadIdx.x;
  if (i < n) out[i] = f2bf(in[i]);
}

// ---------------------------------------------------------------------------
// C[M,N] = A[M,K] @ W[N,K]^T  (bf16 operands, f32 accumulate via WMMA)
// MODE 0: store f32 row-major       (final output projection)
// MODE 1: store bf16 row-major      (Q, K in [B,S,H*dk] layout)
// MODE 2: store bf16 transposed per batch: Vt[(b*N + n)*S + s]  (V^T per head)
// Block: 256 threads (8 waves), 128x128 tile, K-step 32.
// Wave: 64x32 sub-tile = 4x2 fragments of 16x16.
// Mainloop: double-buffered LDS filled by GLOBAL_LOAD_ASYNC_TO_LDS_B128
// (ASYNCcnt-tracked DMA, one tile in flight while WMMAs consume the other).
// ---------------------------------------------------------------------------
template <int MODE>
__global__ __launch_bounds__(256) void gemm_bf16_wmma(
    const u16* __restrict__ A, const u16* __restrict__ W, void* __restrict__ Cout,
    int M, int N, int K, int S)
{
  // 32 cols + 8 pad per row (bank-conflict relief); 2 buffers for pipelining
  __shared__ __align__(16) u16 lsA[2][128 * 40];
  __shared__ __align__(16) u16 lsB[2][128 * 40];

  const int tid  = threadIdx.x;
  const int lane = tid & 31;
  const int wid  = tid >> 5;
  const int wm   = (wid & 1) * 64;   // wave M offset in tile
  const int wn   = (wid >> 1) * 32;  // wave N offset in tile
  const int m0   = blockIdx.y * 128;
  const int n0   = blockIdx.x * 128;
  const int r    = lane & 15;
  const int kh   = (lane >> 4) * 8;  // A/B fragment K sub-chunk select
  const int hi8  = (lane >> 4) * 8;  // C fragment row-half select

  // this thread's two 16B chunks within a 128x32 tile
  const int c0row = tid >> 2,          c0cc = (tid & 3) * 8;        // chunk j=0
  const int c1row = (tid + 256) >> 2,  c1cc = (tid & 3) * 8;        // chunk j=1
  const u32 ldsA0 = (u32)(uintptr_t)&lsA[0][c0row * 40 + c0cc];
  const u32 ldsA1 = (u32)(uintptr_t)&lsA[0][c1row * 40 + c1cc];
  const u32 ldsB0 = (u32)(uintptr_t)&lsB[0][c0row * 40 + c0cc];
  const u32 ldsB1 = (u32)(uintptr_t)&lsB[0][c1row * 40 + c1cc];
  const u32 bufStride = (u32)(128 * 40 * sizeof(u16));

  const v8f vzero = {0.f, 0.f, 0.f, 0.f, 0.f, 0.f, 0.f, 0.f};
  v8f acc[4][2];
#pragma unroll
  for (int mi = 0; mi < 4; ++mi)
#pragma unroll
    for (int ni = 0; ni < 2; ++ni) acc[mi][ni] = vzero;

  // issue one tile's async DMA (4 x b128 per thread: 2 A-chunks + 2 B-chunks)
  auto issue_tile = [&](int k0, int buf) {
    const u32 bo = buf * bufStride;
    const u64 ga0 = (u64)(uintptr_t)(A + (size_t)(m0 + c0row) * K + k0 + c0cc);
    const u64 ga1 = (u64)(uintptr_t)(A + (size_t)(m0 + c1row) * K + k0 + c1cc);
    const u64 gb0 = (u64)(uintptr_t)(W + (size_t)(n0 + c0row) * K + k0 + c0cc);
    const u64 gb1 = (u64)(uintptr_t)(W + (size_t)(n0 + c1row) * K + k0 + c1cc);
    asm volatile("global_load_async_to_lds_b128 %0, %1, off"
                 :: "v"(ldsA0 + bo), "v"(ga0) : "memory");
    asm volatile("global_load_async_to_lds_b128 %0, %1, off"
                 :: "v"(ldsA1 + bo), "v"(ga1) : "memory");
    asm volatile("global_load_async_to_lds_b128 %0, %1, off"
                 :: "v"(ldsB0 + bo), "v"(gb0) : "memory");
    asm volatile("global_load_async_to_lds_b128 %0, %1, off"
                 :: "v"(ldsB1 + bo), "v"(gb1) : "memory");
  };

  const int nsteps = K >> 5;
  issue_tile(0, 0);

  for (int t = 0; t < nsteps; ++t) {
    const int cur = t & 1;
    if (t + 1 < nsteps) {
      issue_tile((t + 1) << 5, cur ^ 1);
      // current tile's 4 loads done; next tile's 4 remain in flight
      asm volatile("s_wait_asynccnt 0x4" ::: "memory");
    } else {
      asm volatile("s_wait_asynccnt 0x0" ::: "memory");
    }
    __syncthreads();   // all waves' DMA for `cur` visible

    // fragment loads (16-bit A-layout: lane r rows, K chunks {kh..kh+7, kh+16..kh+23})
    BF16Frag af[4], bfm[2];
#pragma unroll
    for (int mi = 0; mi < 4; ++mi) {
      const u16* p = &lsA[cur][(wm + mi * 16 + r) * 40 + kh];
      af[mi].u4[0] = *(const uint4*)p;
      af[mi].u4[1] = *(const uint4*)(p + 16);
    }
#pragma unroll
    for (int ni = 0; ni < 2; ++ni) {
      const u16* p = &lsB[cur][(wn + ni * 16 + r) * 40 + kh];
      bfm[ni].u4[0] = *(const uint4*)p;
      bfm[ni].u4[1] = *(const uint4*)(p + 16);
    }

#pragma unroll
    for (int mi = 0; mi < 4; ++mi)
#pragma unroll
      for (int ni = 0; ni < 2; ++ni)
        acc[mi][ni] = __builtin_amdgcn_wmma_f32_16x16x32_bf16(
            false, af[mi].v, false, bfm[ni].v, (short)0, acc[mi][ni], false, false);
    __syncthreads();   // all waves done reading `cur` before it is refilled
  }

  // epilogue (C layout: lane holds column, VGPR v holds row v / v+8)
#pragma unroll
  for (int mi = 0; mi < 4; ++mi) {
#pragma unroll
    for (int ni = 0; ni < 2; ++ni) {
      const int col = n0 + wn + ni * 16 + r;
#pragma unroll
      for (int v = 0; v < 8; ++v) {
        const int row = m0 + wm + mi * 16 + v + hi8;
        const float val = acc[mi][ni][v];
        if (MODE == 0) {
          ((float*)Cout)[(size_t)row * N + col] = val;
        } else if (MODE == 1) {
          ((u16*)Cout)[(size_t)row * N + col] = f2bf(val);
        } else {
          const int b = row / S;
          const int s = row - b * S;
          ((u16*)Cout)[((size_t)(b * N + col)) * S + s] = f2bf(val);
        }
      }
    }
  }
}

// ---------------------------------------------------------------------------
// RoPE (interleaved pairs) applied in place to bf16 Q and K in [B,S,H*dk]
// ---------------------------------------------------------------------------
__global__ void rope_kernel(u16* __restrict__ Qb, u16* __restrict__ Kb, int Bn, int S) {
  const int D = 1024;
  const size_t total = (size_t)Bn * S * (D / 2);
  const size_t t = (size_t)blockIdx.x * blockDim.x + threadIdx.x;
  if (t >= total) return;
  const int i = (int)(t & 31);         // pair index within head (dk/2 = 32)
  const int h = (int)((t >> 5) & 15);
  const size_t bs = t >> 9;
  const int s = (int)(bs % (size_t)S);
  const int b = (int)(bs / (size_t)S);
  const float th = (float)s * __powf(10000.0f, -(float)i * (1.0f / 32.0f));
  float sn, cs;
  __sincosf(th, &sn, &cs);
  const size_t base = ((size_t)b * S + s) * D + h * 64 + 2 * i;
  {
    const float x0 = bf2f(Qb[base]), x1 = bf2f(Qb[base + 1]);
    Qb[base]     = f2bf(cs * x0 - sn * x1);
    Qb[base + 1] = f2bf(sn * x0 + cs * x1);
  }
  {
    const float x0 = bf2f(Kb[base]), x1 = bf2f(Kb[base + 1]);
    Kb[base]     = f2bf(cs * x0 - sn * x1);
    Kb[base + 1] = f2bf(sn * x0 + cs * x1);
  }
}

// ---------------------------------------------------------------------------
// Flash-style causal attention. One wave = one 16-query tile of one (b,h).
// Q fragments resident in registers; K/V B-fragments loaded from global.
// Scores round-trip through per-wave LDS for the C->A layout transpose and
// the online softmax. Vt is [B*H*dk, S] so P@V B-fragments are contiguous.
// ---------------------------------------------------------------------------
__global__ __launch_bounds__(256) void flash_attn_wmma(
    const u16* __restrict__ Qb, const u16* __restrict__ Kb,
    const u16* __restrict__ Vt, u16* __restrict__ Ob, int Bn, int S)
{
  const int D = 1024, H = 16;
  __shared__ __align__(16) float sc[8][16 * 36];   // scores 16x32 (+pad)
  __shared__ __align__(16) u16   sp[8][16 * 40];   // P bf16 16x32 (+pad)
  __shared__ float salp[8][16];                    // per-row broadcast slot

  const int tid  = threadIdx.x;
  const int lane = tid & 31;
  const int w    = tid >> 5;
  const int r    = lane & 15;
  const int hi   = lane >> 4;
  const int kh   = hi * 8;
  const int hi8  = hi * 8;
  const int c0   = hi * 16;

  const int gw = blockIdx.x * 8 + w;
  const int tilesPerSeq = S >> 4;
  const int b   = gw / (H * tilesPerSeq);
  const int rem = gw - b * H * tilesPerSeq;
  const int h   = rem / tilesPerSeq;
  const int q0  = (rem - h * tilesPerSeq) << 4;

  // Q fragments (A layout), dk = 64 -> 2 fragments of 16x32
  BF16Frag qf[2];
  {
    const size_t qbase = ((size_t)b * S + q0 + r) * D + h * 64 + kh;
#pragma unroll
    for (int kf = 0; kf < 2; ++kf) {
      qf[kf].u4[0] = *(const uint4*)(Qb + qbase + kf * 32);
      qf[kf].u4[1] = *(const uint4*)(Qb + qbase + kf * 32 + 16);
    }
  }

  const v8f vzero = {0.f, 0.f, 0.f, 0.f, 0.f, 0.f, 0.f, 0.f};
  v8f ofr[4];
#pragma unroll
  for (int f = 0; f < 4; ++f) ofr[f] = vzero;

  float mrun = -3.0e38f, lrun = 0.f;
  const float scale = 0.125f;  // 1/sqrt(64)
  const int qrow = q0 + r;

  const int ktmax = (q0 + 15) >> 5;
  for (int kt = 0; kt <= ktmax; ++kt) {
    const int kb = kt << 5;

    // S = Q @ K^T for 32 keys (2 N-fragments x 2 K-fragments)
#pragma unroll
    for (int nf = 0; nf < 2; ++nf) {
      v8f s = vzero;
      const size_t krow = ((size_t)b * S + kb + nf * 16 + r) * D + h * 64 + kh;
#pragma unroll
      for (int kf = 0; kf < 2; ++kf) {
        BF16Frag kfr;
        kfr.u4[0] = *(const uint4*)(Kb + krow + kf * 32);
        kfr.u4[1] = *(const uint4*)(Kb + krow + kf * 32 + 16);
        s = __builtin_amdgcn_wmma_f32_16x16x32_bf16(
            false, qf[kf].v, false, kfr.v, (short)0, s, false, false);
      }
#pragma unroll
      for (int v = 0; v < 8; ++v)
        sc[w][(v + hi8) * 36 + nf * 16 + r] = s[v];
    }
    asm volatile("s_wait_dscnt 0" ::: "memory");

    // online softmax: lane handles row r, columns c0..c0+15
    float pv[16];
#pragma unroll
    for (int j = 0; j < 4; ++j) {
      const float4 f4 = *(const float4*)&sc[w][r * 36 + c0 + j * 4];
      pv[j * 4 + 0] = f4.x; pv[j * 4 + 1] = f4.y;
      pv[j * 4 + 2] = f4.z; pv[j * 4 + 3] = f4.w;
    }
    float tmax = -3.0e38f;
#pragma unroll
    for (int j = 0; j < 16; ++j) {
      const int key = kb + c0 + j;
      const float vsc = (key <= qrow) ? pv[j] * scale : -3.0e38f;  // causal mask
      pv[j] = vsc;
      tmax = fmaxf(tmax, vsc);
    }
    tmax = fmaxf(tmax, __shfl_xor(tmax, 16));
    const float mnew  = fmaxf(mrun, tmax);
    const float alpha = __expf(mrun - mnew);
    float psum = 0.f;
#pragma unroll
    for (int j = 0; j < 16; j += 2) {
      const float p0 = __expf(pv[j] - mnew);
      const float p1 = __expf(pv[j + 1] - mnew);
      psum += p0 + p1;
      const u32 pk = (u32)f2bf(p0) | ((u32)f2bf(p1) << 16);
      *(u32*)&sp[w][r * 40 + c0 + j] = pk;
    }
    psum += __shfl_xor(psum, 16);
    lrun = alpha * lrun + psum;
    mrun = mnew;
    salp[w][r] = alpha;  // both halves write identical value
    asm volatile("s_wait_dscnt 0" ::: "memory");

    // rescale running O by alpha (per-row broadcast via LDS)
#pragma unroll
    for (int v = 0; v < 8; ++v) {
      const float av = salp[w][v + hi8];
#pragma unroll
      for (int f = 0; f < 4; ++f) ofr[f][v] *= av;
    }

    // P fragment (A layout, 16x32)
    BF16Frag pf;
    pf.u4[0] = *(const uint4*)&sp[w][r * 40 + kh];
    pf.u4[1] = *(const uint4*)&sp[w][r * 40 + kh + 16];

    // O += P @ V   (V^T rows contiguous along keys)
#pragma unroll
    for (int nf = 0; nf < 4; ++nf) {
      const size_t vtrow = ((size_t)(b * 1024 + h * 64 + nf * 16 + r)) * S + kb + kh;
      BF16Frag vfr;
      vfr.u4[0] = *(const uint4*)(Vt + vtrow);
      vfr.u4[1] = *(const uint4*)(Vt + vtrow + 16);
      ofr[nf] = __builtin_amdgcn_wmma_f32_16x16x32_bf16(
          false, pf.v, false, vfr.v, (short)0, ofr[nf], false, false);
    }
  }

  // finalize: divide by l, store bf16 merged-head output [B,S,H*dk]
  salp[w][r] = 1.0f / lrun;
  asm volatile("s_wait_dscnt 0" ::: "memory");
#pragma unroll
  for (int v = 0; v < 8; ++v) {
    const float inv = salp[w][v + hi8];
    const int row = q0 + v + hi8;
#pragma unroll
    for (int nf = 0; nf < 4; ++nf) {
      const size_t oaddr = ((size_t)b * S + row) * D + h * 64 + nf * 16 + r;
      Ob[oaddr] = f2bf(ofr[nf][v] * inv);
    }
  }
}

// ---------------------------------------------------------------------------
// Host-side orchestration
// ---------------------------------------------------------------------------
extern "C" void kernel_launch(void* const* d_in, const int* in_sizes, int n_in,
                              void* d_out, int out_size, void* d_ws, size_t ws_size,
                              hipStream_t stream) {
  (void)in_sizes; (void)n_in; (void)out_size; (void)ws_size;
  const float* x  = (const float*)d_in[0];
  const float* WQ = (const float*)d_in[1];
  const float* WK = (const float*)d_in[2];
  const float* WV = (const float*)d_in[3];
  const float* WO = (const float*)d_in[4];
  float* out = (float*)d_out;

  const int B = 4, S = 2048, D = 1024;
  const int M = B * S;                 // 8192
  const size_t MD = (size_t)M * D;     // 8Mi
  const size_t DD = (size_t)D * D;     // 1Mi

  // workspace layout (bf16 buffers): 88 MB total
  u16* xb = (u16*)d_ws;
  u16* wq = xb + MD;
  u16* wk = wq + DD;
  u16* wv = wk + DD;
  u16* wo = wv + DD;
  u16* Qb = wo + DD;
  u16* Kb = Qb + MD;
  u16* Vt = Kb + MD;
  u16* Ob = Vt + MD;

  // 1) casts
  cast_f32_to_bf16<<<(unsigned)((MD + 255) / 256), 256, 0, stream>>>(x, xb, MD);
  cast_f32_to_bf16<<<(unsigned)((DD + 255) / 256), 256, 0, stream>>>(WQ, wq, DD);
  cast_f32_to_bf16<<<(unsigned)((DD + 255) / 256), 256, 0, stream>>>(WK, wk, DD);
  cast_f32_to_bf16<<<(unsigned)((DD + 255) / 256), 256, 0, stream>>>(WV, wv, DD);
  cast_f32_to_bf16<<<(unsigned)((DD + 255) / 256), 256, 0, stream>>>(WO, wo, DD);

  // 2) Q/K/V projections (WMMA GEMMs, async-DMA double-buffered)
  dim3 ggrid(D / 128, M / 128);
  gemm_bf16_wmma<1><<<ggrid, 256, 0, stream>>>(xb, wq, Qb, M, D, D, S);
  gemm_bf16_wmma<1><<<ggrid, 256, 0, stream>>>(xb, wk, Kb, M, D, D, S);
  gemm_bf16_wmma<2><<<ggrid, 256, 0, stream>>>(xb, wv, Vt, M, D, D, S);

  // 3) RoPE on Q, K
  const size_t pairs = MD / 2;
  rope_kernel<<<(unsigned)((pairs + 255) / 256), 256, 0, stream>>>(Qb, Kb, B, S);

  // 4) flash attention (8192 query tiles / 8 waves per block)
  flash_attn_wmma<<<(B * 16 * (S / 16)) / 8, 256, 0, stream>>>(Qb, Kb, Vt, Ob, B, S);

  // 5) output projection, fp32 result
  gemm_bf16_wmma<0><<<ggrid, 256, 0, stream>>>(Ob, wo, out, M, D, D, S);
}
